// GridRnn_5557687681795
// MI455X (gfx1250) — matrix operand order repaired and checked by reading, not compiled, not run on hardware
//
#include <hip/hip_runtime.h>
#include <math.h>

// ---------------- problem constants ----------------
#define HIDN  512
#define EMBD  512
#define BSZN  4096
#define CN    4
#define LN    4
#define NHN   8
#define DHN   64
#define OUTN  32000
#define G3H   1536          // 3*HID

#define CB    ((long)CN * BSZN)          // 16384 rows
#define CBH   ((long)CB * HIDN)          // 8,388,608
#define NKW   ((long)G3H * HIDN)         // 786,432  (one 1536x512 weight)
#define NKO   ((long)HIDN * HIDN)        // 262,144
#define NKH   ((long)OUTN * HIDN)        // 16,384,000

typedef __attribute__((ext_vector_type(16))) __bf16 v16bf;
typedef __attribute__((ext_vector_type(8)))  __bf16 v8bf;
typedef __attribute__((ext_vector_type(8)))  float  v8f;

__device__ __forceinline__ float sigm(float x) { return 1.0f / (1.0f + expf(-x)); }

// ---------------- weight packer: fp32 row-major (B,N,K) -> bf16 WMMA B-fragments ----
// Per 32x16 (KxN) tile: lane = (n&15) + 16*((k32>>3)&1), half index i = (k32&7) + 8*(k32>>4) (table K order).
__global__ void k_pack_w(const float* __restrict__ W, __bf16* __restrict__ P,
                         int N, int K, long total)
{
    long idx = (long)blockIdx.x * 256 + threadIdx.x;
    if (idx >= total) return;
    long nk = (long)N * K;
    long b  = idx / nk;
    long r  = idx - b * nk;
    int  n  = (int)(r / K);
    int  k  = (int)(r - (long)n * K);
    int  t   = n >> 4;
    int  s   = k >> 5;
    int  k32 = k & 31;
    int  lane = (n & 15) + (((k32 >> 3) & 1) << 4);
    int  i    = (k32 & 7) + ((k32 >> 4) << 3);
    int  KT   = K >> 5;
    long dst = b * nk + (((((long)t * KT + s) * 32) + lane) << 4) + i;
    P[dst] = (__bf16)W[idx];
}

// ---------------- generic fp32 -> bf16 cast ----------------
__global__ void k_cast(const float* __restrict__ in, __bf16* __restrict__ out, long n)
{
    long idx = (long)blockIdx.x * 256 + threadIdx.x;
    if (idx < n) out[idx] = (__bf16)in[idx];
}

// ---------------- embedding gather -> bf16 ----------------
__global__ void k_embed(const int* __restrict__ tokens, const float* __restrict__ emb,
                        __bf16* __restrict__ x)
{
    long idx = (long)blockIdx.x * 256 + threadIdx.x;
    if (idx >= (long)BSZN * EMBD) return;
    int b = (int)(idx >> 9);           // /512
    int j = (int)(idx & 511);
    x[idx] = (__bf16)emb[(long)tokens[b] * EMBD + j];
}

// ---------------- layer-0 cols 1..3: gx = b_ih_0r broadcast ----------------
__global__ void k_fill_gx0(const float* __restrict__ b_ih_0r, float* __restrict__ gx)
{
    long idx = (long)blockIdx.x * 256 + threadIdx.x;
    long total = 3L * BSZN * G3H;
    if (idx >= total) return;
    int  j  = (int)(idx % G3H);
    int  c1 = (int)(idx / ((long)BSZN * G3H));    // 0..2  -> column c1+1
    gx[(long)BSZN * G3H + idx] = b_ih_0r[c1 * G3H + j];
}

// ---------------- WMMA bf16 GEMM:  C[M,N] = A[M,K] * W^T + bias ----------------
// A: bf16 row-major (lda=K).  P: packed bf16 B-fragments of W (N_w x K), tile offset nt0.
// grid = (N/128, M/128), block = 256 threads = 8 waves (4 M-groups x 2 N-groups);
// each wave computes a 32x64 strip: 2 A-fragments x 4 B-fragments = 8 WMMA per K-step.
// Manual 2x ping-pong unroll with disjoint register sets (X/Y): loads land directly in
// WMMA operand registers (no rotation copies, partial s_wait_loadcnt only).
__global__ __launch_bounds__(256) void k_gemm(
    const __bf16* __restrict__ A, const __bf16* __restrict__ P,
    const float* __restrict__ bias, float* __restrict__ Cout,
    int N, int K, int nt0)
{
    const int lane = threadIdx.x & 31;
    const int wave = threadIdx.x >> 5;
    const int wm   = wave & 3;
    const int wn   = wave >> 2;
    const int row0 = blockIdx.y * 128 + wm * 32;
    const int col0 = blockIdx.x * 128 + wn * 64;
    const int m    = lane & 15;
    const int kh   = lane >> 4;
    const int KT   = K >> 5;      // 16 for K=512 (even; ping-pong assumes KT >= 4, even)

    v8f acc[2][4];
#pragma unroll
    for (int im = 0; im < 2; ++im)
#pragma unroll
        for (int j = 0; j < 4; ++j)
#pragma unroll
            for (int e = 0; e < 8; ++e) acc[im][j][e] = 0.0f;

    // 16-bit A 16x32 layout: per-lane half-runs at K = kh*8 and K = 16+kh*8
    const __bf16* arow0 = A + (long)(row0 + m) * K + kh * 8;
    const __bf16* arow1 = arow0 + (long)16 * K;
    const __bf16* pj[4];
#pragma unroll
    for (int j = 0; j < 4; ++j)
        pj[j] = P + ((((long)(nt0 + (col0 >> 4) + j) * KT) * 32 + lane) << 4);

    v8bf Xal0, Xah0, Xal1, Xah1;  v16bf Xb0, Xb1, Xb2, Xb3;
    v8bf Yal0, Yah0, Yal1, Yah1;  v16bf Yb0, Yb1, Yb2, Yb3;

#define LOADF(P_, S_) do {                                                      \
        const int kb_ = (S_) << 5;                                              \
        P_##al0 = *(const v8bf*)(arow0 + kb_);                                  \
        P_##ah0 = *(const v8bf*)(arow0 + kb_ + 16);                             \
        P_##al1 = *(const v8bf*)(arow1 + kb_);                                  \
        P_##ah1 = *(const v8bf*)(arow1 + kb_ + 16);                             \
        P_##b0  = *(const v16bf*)(pj[0] + ((long)(S_) << 9));                   \
        P_##b1  = *(const v16bf*)(pj[1] + ((long)(S_) << 9));                   \
        P_##b2  = *(const v16bf*)(pj[2] + ((long)(S_) << 9));                   \
        P_##b3  = *(const v16bf*)(pj[3] + ((long)(S_) << 9));                   \
    } while (0)

#define WMMAF(P_) do {                                                          \
        v16bf af0_, af1_;                                                       \
        _Pragma("unroll")                                                       \
        for (int i_ = 0; i_ < 8; ++i_) {                                        \
            af0_[i_] = P_##al0[i_]; af0_[8 + i_] = P_##ah0[i_];                 \
            af1_[i_] = P_##al1[i_]; af1_[8 + i_] = P_##ah1[i_];                 \
        }                                                                       \
        acc[0][0] = __builtin_amdgcn_wmma_f32_16x16x32_bf16(                    \
            false, af0_, false, P_##b0, (short)0, acc[0][0], false, false);     \
        acc[1][0] = __builtin_amdgcn_wmma_f32_16x16x32_bf16(                    \
            false, af1_, false, P_##b0, (short)0, acc[1][0], false, false);     \
        acc[0][1] = __builtin_amdgcn_wmma_f32_16x16x32_bf16(                    \
            false, af0_, false, P_##b1, (short)0, acc[0][1], false, false);     \
        acc[1][1] = __builtin_amdgcn_wmma_f32_16x16x32_bf16(                    \
            false, af1_, false, P_##b1, (short)0, acc[1][1], false, false);     \
        acc[0][2] = __builtin_amdgcn_wmma_f32_16x16x32_bf16(                    \
            false, af0_, false, P_##b2, (short)0, acc[0][2], false, false);     \
        acc[1][2] = __builtin_amdgcn_wmma_f32_16x16x32_bf16(                    \
            false, af1_, false, P_##b2, (short)0, acc[1][2], false, false);     \
        acc[0][3] = __builtin_amdgcn_wmma_f32_16x16x32_bf16(                    \
            false, af0_, false, P_##b3, (short)0, acc[0][3], false, false);     \
        acc[1][3] = __builtin_amdgcn_wmma_f32_16x16x32_bf16(                    \
            false, af1_, false, P_##b3, (short)0, acc[1][3], false, false);     \
    } while (0)

    LOADF(X, 0);
    int s = 0;
    for (; s + 2 < KT; s += 2) {
        LOADF(Y, s + 1);
        __builtin_prefetch(arow0 + ((s + 2) << 5) + 32, 0, 1);  // global_prefetch_b8
        WMMAF(X);
        LOADF(X, s + 2);
        WMMAF(Y);
    }
    LOADF(Y, KT - 1);
    WMMAF(X);
    WMMAF(Y);

#undef LOADF
#undef WMMAF

#pragma unroll
    for (int im = 0; im < 2; ++im)
#pragma unroll
        for (int j = 0; j < 4; ++j) {
            const int n = col0 + j * 16 + m;
            const float bv = bias[n];
#pragma unroll
            for (int r = 0; r < 8; ++r) {
                const int mm = row0 + im * 16 + kh * 8 + r;  // C/D: M = r + 8*(lane>>4)
                Cout[(long)mm * N + n] = acc[im][j][r] + bv;
            }
        }
}

// ---------------- GRU pointwise ----------------
// gx, gh: (CB, 1536) already include biases.  hold: (CB, 512) fp32.
__global__ void k_gru(const float* __restrict__ gx, const float* __restrict__ gh,
                      const float* __restrict__ hold, const float* __restrict__ ids_l,
                      float* __restrict__ hgru_f, __bf16* __restrict__ hgru_b,
                      __bf16* __restrict__ qk_b)
{
    long idx = (long)blockIdx.x * 256 + threadIdx.x;
    if (idx >= CBH) return;
    int  j  = (int)(idx & (HIDN - 1));
    long cb = idx >> 9;
    int  c  = (int)(cb >> 12);                  // /BSZ
    long g  = cb * G3H;
    float r = sigm(gx[g + j]            + gh[g + j]);
    float z = sigm(gx[g + HIDN + j]     + gh[g + HIDN + j]);
    float nn = tanhf(gx[g + 2*HIDN + j] + r * gh[g + 2*HIDN + j]);
    float hnew = (1.0f - z) * nn + z * hold[idx];
    hgru_f[idx] = hnew;
    hgru_b[idx] = (__bf16)hnew;
    qk_b[idx]   = (__bf16)(hnew + ids_l[c * HIDN + j]);
}

// ---------------- tiny attention over C=4 positions ----------------
// qkproj: (CB, 1024) rows c*BSZ+b, cols [0,512)=q, [512,1024)=k.  v: (CB,512).  o: bf16 (CB,512)
__global__ __launch_bounds__(256) void k_attn(const float* __restrict__ qk,
                                              const float* __restrict__ v,
                                              __bf16* __restrict__ o)
{
    int idx = blockIdx.x * 256 + threadIdx.x;
    if (idx >= BSZN * NHN * CN) return;
    int qc = idx & 3;
    int nh = (idx >> 2) & 7;
    int b  = idx >> 5;
    const float* qrow = qk + ((long)qc * BSZN + b) * 1024 + nh * DHN;
    const float* krow[CN];
    const float* vrow[CN];
#pragma unroll
    for (int c = 0; c < CN; ++c) {
        krow[c] = qk + ((long)c * BSZN + b) * 1024 + 512 + nh * DHN;
        vrow[c] = v  + ((long)c * BSZN + b) * 512 + nh * DHN;
    }
    float s[CN];
    float mx = -1e30f;
#pragma unroll
    for (int c = 0; c < CN; ++c) {
        float a = 0.0f;
        for (int d = 0; d < DHN; ++d) a += qrow[d] * krow[c][d];
        s[c] = a * 0.125f;              // 1/sqrt(64)
        mx = fmaxf(mx, s[c]);
    }
    float den = 0.0f;
#pragma unroll
    for (int c = 0; c < CN; ++c) { s[c] = expf(s[c] - mx); den += s[c]; }
    float inv = 1.0f / den;
    __bf16* orow = o + ((long)qc * BSZN + b) * 512 + nh * DHN;
    for (int d = 0; d < DHN; ++d) {
        float a = 0.0f;
#pragma unroll
        for (int c = 0; c < CN; ++c) a += s[c] * vrow[c][d];
        orow[d] = (__bf16)(a * inv);
    }
}

// ---------------- fused LayerNorm + scalar gate + state update ----------------
// one block per row (c*BSZ+b): LN(msg0) -> msg; g = sigm([hgru,msg]·gw + gb);
// hout = (1-g)*hgru + g*msg  -> fp32 (h_n output) + bf16 (xin for next layer / head)
__global__ __launch_bounds__(256) void k_lngate(
    const float* __restrict__ msg0, const float* __restrict__ hgru,
    const float* __restrict__ lng, const float* __restrict__ lnb,
    const float* __restrict__ gw, const float* __restrict__ gb,
    float* __restrict__ hout, __bf16* __restrict__ xout)
{
    const int row = blockIdx.x;
    const int t   = threadIdx.x;
    const long base = (long)row * HIDN;
    float m0 = msg0[base + t];
    float m1 = msg0[base + 256 + t];
    __shared__ float s1[256];
    __shared__ float s2[256];
    s1[t] = m0 + m1;
    s2[t] = m0 * m0 + m1 * m1;
    __syncthreads();
    for (int off = 128; off > 0; off >>= 1) {
        if (t < off) { s1[t] += s1[t + off]; s2[t] += s2[t + off]; }
        __syncthreads();
    }
    float mu   = s1[0] * (1.0f / HIDN);
    float var  = s2[0] * (1.0f / HIDN) - mu * mu;
    float rstd = rsqrtf(var + 1e-5f);
    __syncthreads();
    float ln0 = (m0 - mu) * rstd * lng[t]       + lnb[t];
    float ln1 = (m1 - mu) * rstd * lng[256 + t] + lnb[256 + t];
    float h0 = hgru[base + t];
    float h1 = hgru[base + 256 + t];
    s1[t] = h0 * gw[t] + h1 * gw[256 + t] + ln0 * gw[512 + t] + ln1 * gw[768 + t];
    __syncthreads();
    for (int off = 128; off > 0; off >>= 1) {
        if (t < off) s1[t] += s1[t + off];
        __syncthreads();
    }
    float g = sigm(s1[0] + gb[0]);
    float o0 = (1.0f - g) * h0 + g * ln0;
    float o1 = (1.0f - g) * h1 + g * ln1;
    hout[base + t]       = o0;
    hout[base + 256 + t] = o1;
    xout[base + t]       = (__bf16)o0;
    xout[base + 256 + t] = (__bf16)o1;
}

// ---------------- host orchestration ----------------
#define L1D(kern, total, ...)                                                   \
    do {                                                                        \
        long _t = (long)(total);                                                \
        unsigned _b = (unsigned)((_t + 255) / 256);                             \
        kern<<<dim3(_b), dim3(256), 0, stream>>>(__VA_ARGS__);                  \
    } while (0)

extern "C" void kernel_launch(void* const* d_in, const int* in_sizes, int n_in,
                              void* d_out, int out_size, void* d_ws, size_t ws_size,
                              hipStream_t stream)
{
    (void)in_sizes; (void)n_in; (void)out_size; (void)ws_size;

    const int*   tokens  = (const int*)  d_in[0];
    const float* h       = (const float*)d_in[1];
    const float* emb     = (const float*)d_in[2];
    const float* W_ih_00 = (const float*)d_in[3];
    const float* W_hh_00 = (const float*)d_in[4];
    const float* b_ih_00 = (const float*)d_in[5];
    const float* b_hh_00 = (const float*)d_in[6];
    /* d_in[7] = W_ih_0r multiplies an all-zero input -> unused */
    const float* W_hh_0r = (const float*)d_in[8];
    const float* b_ih_0r = (const float*)d_in[9];
    const float* b_hh_0r = (const float*)d_in[10];
    const float* W_ih    = (const float*)d_in[11];
    const float* W_hh    = (const float*)d_in[12];
    const float* b_ih    = (const float*)d_in[13];
    const float* b_hh    = (const float*)d_in[14];
    const float* ids     = (const float*)d_in[15];
    const float* in_w    = (const float*)d_in[16];
    const float* in_b    = (const float*)d_in[17];
    const float* out_w   = (const float*)d_in[18];
    const float* out_b   = (const float*)d_in[19];
    const float* ln_g    = (const float*)d_in[20];
    const float* ln_b    = (const float*)d_in[21];
    const float* gate_w  = (const float*)d_in[22];
    const float* gate_b  = (const float*)d_in[23];
    const float* head_w  = (const float*)d_in[24];
    const float* head_b  = (const float*)d_in[25];

    float* y  = (float*)d_out;                       // (4096, 32000)
    float* hn = y + (long)BSZN * OUTN;               // (L, C, BSZ, HID)

    // --- workspace carve-out ---
    char* ws = (char*)d_ws;
    size_t off = 0;
    auto carve = [&](size_t bytes) -> void* {
        void* p = (void*)(ws + off);
        off = (off + bytes + 255) & ~(size_t)255;
        return p;
    };
    __bf16* P_ih00 = (__bf16*)carve(NKW * 2);
    __bf16* P_hh00 = (__bf16*)carve(NKW * 2);
    __bf16* P_hh0r = (__bf16*)carve(3 * NKW * 2);
    __bf16* P_ih   = (__bf16*)carve(12 * NKW * 2);
    __bf16* P_hh   = (__bf16*)carve(12 * NKW * 2);
    __bf16* P_inw  = (__bf16*)carve(4 * NKW * 2);
    __bf16* P_outw = (__bf16*)carve(4 * NKO * 2);
    __bf16* P_head = (__bf16*)carve(NKH * 2);
    __bf16* x_b    = (__bf16*)carve((long)BSZN * EMBD * 2);
    __bf16* hold_b = (__bf16*)carve(CBH * 2);
    float*  gx     = (float*) carve(CB * G3H * 4);
    float*  gh     = (float*) carve(CB * G3H * 4);
    float*  hgru_f = (float*) carve(CBH * 4);
    __bf16* hgru_b = (__bf16*)carve(CBH * 2);
    __bf16* qk_b   = (__bf16*)carve(CBH * 2);
    float*  qkproj = (float*) carve(CB * 1024 * 4);
    float*  vbuf   = (float*) carve(CBH * 4);
    __bf16* obuf   = (__bf16*)carve(CBH * 2);
    float*  msg0   = (float*) carve(CBH * 4);
    __bf16* xin_b  = (__bf16*)carve(CBH * 2);

    auto gemm = [&](const __bf16* Ap, const __bf16* Pp, const float* bp,
                    float* Cp, int M, int N, int nt0) {
        dim3 grid(N / 128, M / 128);
        k_gemm<<<grid, dim3(256), 0, stream>>>(Ap, Pp, bp, Cp, N, HIDN, nt0);
    };

    // --- pack weights to bf16 WMMA fragments (must rerun each call: no caching allowed) ---
    L1D(k_pack_w, NKW,      W_ih_00, P_ih00, G3H,  HIDN, NKW);
    L1D(k_pack_w, NKW,      W_hh_00, P_hh00, G3H,  HIDN, NKW);
    L1D(k_pack_w, 3 * NKW,  W_hh_0r, P_hh0r, G3H,  HIDN, 3 * NKW);
    L1D(k_pack_w, 12 * NKW, W_ih,    P_ih,   G3H,  HIDN, 12 * NKW);
    L1D(k_pack_w, 12 * NKW, W_hh,    P_hh,   G3H,  HIDN, 12 * NKW);
    L1D(k_pack_w, 4 * NKW,  in_w,    P_inw,  G3H,  HIDN, 4 * NKW);
    L1D(k_pack_w, 4 * NKO,  out_w,   P_outw, HIDN, HIDN, 4 * NKO);
    L1D(k_pack_w, NKH,      head_w,  P_head, OUTN, HIDN, NKH);

    // --- embedding gather ---
    L1D(k_embed, (long)BSZN * EMBD, tokens, emb, x_b);

    for (int l = 0; l < LN; ++l) {
        // cast this layer's initial hidden state to bf16
        L1D(k_cast, CBH, h + (long)l * CBH, hold_b, CBH);

        if (l == 0) {
            gemm(x_b, P_ih00, b_ih_00, gx, BSZN, G3H, 0);
            L1D(k_fill_gx0, 3L * BSZN * G3H, b_ih_0r, gx);
            gemm(hold_b, P_hh00, b_hh_00, gh, BSZN, G3H, 0);
            for (int c = 1; c < CN; ++c)
                gemm(hold_b + (long)c * BSZN * HIDN,
                     P_hh0r + (long)(c - 1) * NKW,
                     b_hh_0r + (long)(c - 1) * G3H,
                     gh + (long)c * BSZN * G3H, BSZN, G3H, 0);
        } else {
            for (int c = 0; c < CN; ++c) {
                long wi = (long)(l - 1) * CN + c;
                gemm(xin_b + (long)c * BSZN * HIDN,
                     P_ih + wi * NKW, b_ih + wi * G3H,
                     gx + (long)c * BSZN * G3H, BSZN, G3H, 0);
                gemm(hold_b + (long)c * BSZN * HIDN,
                     P_hh + wi * NKW, b_hh + wi * G3H,
                     gh + (long)c * BSZN * G3H, BSZN, G3H, 0);
            }
        }

        // GRU pointwise: h_gru (+ids -> qk)
        L1D(k_gru, CBH, gx, gh, h + (long)l * CBH, ids + (long)l * CN * HIDN,
            hgru_f, hgru_b, qk_b);

        // q,k projection (N=1024 = rows 0..1023 of in_w[l]); v projection (rows 1024..1535)
        gemm(qk_b,   P_inw + (long)l * NKW, in_b + (long)l * G3H,        qkproj, (int)CB, 1024, 0);
        gemm(hgru_b, P_inw + (long)l * NKW, in_b + (long)l * G3H + 1024, vbuf,   (int)CB, 512, 64);

        // attention over the C=4 columns
        L1D(k_attn, (long)BSZN * NHN * CN, qkproj, vbuf, obuf);

        // output projection
        gemm(obuf, P_outw + (long)l * NKO, out_b + (long)l * HIDN, msg0, (int)CB, 512, 0);

        // LN + gate + update -> h_n output (fp32) and next-layer input (bf16)
        k_lngate<<<dim3((unsigned)CB), dim3(256), 0, stream>>>(
            msg0, hgru_f, ln_g + (long)l * HIDN, ln_b + (long)l * HIDN,
            gate_w + (long)l * 2 * HIDN, gate_b + l,
            hn + (long)l * CBH, xin_b);
    }

    // LM head: rows 0..4095 of xin_b are column 0 of the last layer
    gemm(xin_b, P_head, head_b, y, BSZN, OUTN, 0);
}
